// NemotronHMoE_52716428591170
// MI455X (gfx1250) — matrix operand order repaired and checked by reading, not compiled, not run on hardware
//
#include <hip/hip_runtime.h>
#include <hip/hip_bf16.h>

typedef __bf16 bf16;
typedef __attribute__((ext_vector_type(16))) __bf16 v16bf;
typedef __attribute__((ext_vector_type(8)))  float  v8f;

#define T_TOK  4096
#define H_DIM  1024
#define E_NUM  32
#define I_DIM  512
#define SI_DIM 2048
#define TOPK   8
#define NGRP   8
#define TOPG   4
#define RSCALE 2.5f

// ------------- gfx1250 async global->LDS copy (ASYNCcnt path) -------------

#if defined(__gfx1250__) && __has_builtin(__builtin_amdgcn_global_load_async_to_lds_b128) && \
    __has_builtin(__builtin_amdgcn_s_wait_asynccnt)
#define USE_ASYNC_LDS 1
typedef int i32x4 __attribute__((ext_vector_type(4)));
typedef __attribute__((address_space(1))) i32x4 g_i32x4;  // global
typedef __attribute__((address_space(3))) i32x4 l_i32x4;  // LDS
#else
#define USE_ASYNC_LDS 0
#endif

// copy one uint4 (16B) from global to LDS, per-lane
__device__ __forceinline__ void copy16_g2l(uint4* dst_lds, const uint4* src_glb) {
#if USE_ASYNC_LDS
    __builtin_amdgcn_global_load_async_to_lds_b128((g_i32x4*)src_glb,
                                                   (l_i32x4*)dst_lds, 0, 0);
#else
    *dst_lds = *src_glb;
#endif
}

__device__ __forceinline__ void copy_fence_g2l() {
#if USE_ASYNC_LDS
    __builtin_amdgcn_s_wait_asynccnt(0);
#endif
}

// ---------------- WMMA helpers (CDNA5 V_WMMA_F32_16X16X32_BF16) ----------------

__device__ __forceinline__ v8f wmma_bf16(v16bf a, v16bf b, v8f c) {
    // 8 args: (neg_a, A, neg_b, B, c_mod, C, reuse_a, reuse_b)
    return __builtin_amdgcn_wmma_f32_16x16x32_bf16(false, a, false, b, (short)0, c,
                                                   false, false);
}

// A-matrix fragment (16x32 bf16, row-major in LDS with leading dim `ld`).
// Layout used here (contiguous k): lane L holds row M = L%16; element e maps to
//   K = kbase + (L>=16 ? 16 : 0) + e
__device__ __forceinline__ v16bf load_a_frag(const bf16* __restrict__ base, int ld,
                                             int mbase, int kbase, int lane) {
    const int m    = mbase + (lane & 15);
    const int koff = kbase + (lane & 16);          // +16 for upper half-wave
    const bf16* p = base + m * ld + koff;
    v16bf a;
#pragma unroll
    for (int e = 0; e < 8; ++e) a[e] = p[e];          // K = koff + 0..7
#pragma unroll
    for (int e = 0; e < 8; ++e) a[e + 8] = p[e + 8];  // K = koff + 8..15
    return a;
}

// B-matrix fragment (32x16): B[k][n] = W[n][k], W row-major fp32 [N][Kld].
// Layout used here (contiguous k): lane L holds col N = L%16; element e maps to
//   K = kbase + (L>=16 ? 16 : 0) + e   -> 16 consecutive fp32 (64B) per lane.
__device__ __forceinline__ v16bf load_b_frag(const float* __restrict__ W, int ldk,
                                             int nbase, int kbase, int lane) {
    const int n  = nbase + (lane & 15);
    const int k0 = kbase + (lane & 16);
    const float* p = W + (size_t)n * ldk + k0;
    const float4* q = (const float4*)p;
    float4 f0 = q[0], f1 = q[1], f2 = q[2], f3 = q[3];
    __builtin_prefetch(p + 32, 0, 1);   // next K-step -> global_prefetch_b8
    v16bf b;
    b[0]  = (bf16)f0.x; b[1]  = (bf16)f0.y; b[2]  = (bf16)f0.z; b[3]  = (bf16)f0.w;
    b[4]  = (bf16)f1.x; b[5]  = (bf16)f1.y; b[6]  = (bf16)f1.z; b[7]  = (bf16)f1.w;
    b[8]  = (bf16)f2.x; b[9]  = (bf16)f2.y; b[10] = (bf16)f2.z; b[11] = (bf16)f2.w;
    b[12] = (bf16)f3.x; b[13] = (bf16)f3.y; b[14] = (bf16)f3.z; b[15] = (bf16)f3.w;
    return b;
}

__device__ __forceinline__ void zero8(v8f& v) {
#pragma unroll
    for (int i = 0; i < 8; ++i) v[i] = 0.0f;
}

// GEMM over a chunk of 4 column tiles (64 cols): A frags loaded once per K-step,
// reused by 8 WMMAs. acc[m][j]: m = M-tile (rows 0-15 / 16-31), j = col tile.
__device__ __forceinline__ void mm_chunk4(const bf16* __restrict__ A, int lda,
                                          const float* __restrict__ W, int ldw,
                                          int kdim, int ncol0, int lane,
                                          v8f acc[2][4]) {
#pragma unroll
    for (int m = 0; m < 2; ++m)
#pragma unroll
        for (int j = 0; j < 4; ++j) zero8(acc[m][j]);
    for (int kk = 0; kk < kdim; kk += 32) {
        v16bf a0 = load_a_frag(A, lda, 0, kk, lane);
        v16bf a1 = load_a_frag(A, lda, 16, kk, lane);
#pragma unroll
        for (int j = 0; j < 4; ++j) {
            v16bf b = load_b_frag(W, ldw, ncol0 + 16 * j, kk, lane);
            acc[0][j] = wmma_bf16(a0, b, acc[0][j]);
            acc[1][j] = wmma_bf16(a1, b, acc[1][j]);
        }
    }
}

// ---------------- kernel 0: zero per-expert counters ----------------

__global__ void init_cnt_kernel(int* __restrict__ cnt) {
    if (threadIdx.x < E_NUM) cnt[threadIdx.x] = 0;
}

// ---------------- kernel 1: router (grouped top-k, sigmoid + bias) ----------------
// One block (256 thr) per token. Also converts x -> bf16 workspace copy.

__global__ void __launch_bounds__(256)
router_kernel(const float* __restrict__ x, const float* __restrict__ gate_w,
              const float* __restrict__ bias, bf16* __restrict__ xb,
              int* __restrict__ cnt, int* __restrict__ perm,
              float* __restrict__ pgate) {
    __shared__ float xs[H_DIM];
    __shared__ float red[8][E_NUM];
    __shared__ float sf[E_NUM], sc[E_NUM], gs[NGRP], mk[E_NUM];
    __shared__ int   gsel[NGRP], sel[E_NUM];

    const int t   = blockIdx.x;
    const int tid = threadIdx.x;

    for (int i = tid; i < H_DIM; i += 256) {
        float v = x[(size_t)t * H_DIM + i];
        xs[i] = v;
        xb[(size_t)t * H_DIM + i] = (bf16)v;
    }
    __syncthreads();

    {   // logits: thread (e = tid&31) x (chunk c = tid>>5) partial dots
        const int e = tid & 31, c = tid >> 5;
        const float* w  = gate_w + (size_t)e * H_DIM + c * 128;
        const float* xv = xs + c * 128;
        float p = 0.f;
#pragma unroll 8
        for (int j = 0; j < 128; ++j) p += xv[j] * w[j];
        red[c][e] = p;
    }
    __syncthreads();

    if (tid < E_NUM) {
        float l = 0.f;
#pragma unroll
        for (int c = 0; c < 8; ++c) l += red[c][tid];
        float s = 1.f / (1.f + __expf(-l));
        sc[tid] = s;              // raw score (weights)
        sf[tid] = s + bias[tid];  // score-for-choice
    }
    __syncthreads();

    if (tid < NGRP) {  // sum of top-2 inside each group of 4
        float a0 = sf[4 * tid], a1 = sf[4 * tid + 1];
        float a2 = sf[4 * tid + 2], a3 = sf[4 * tid + 3];
        float m01 = fmaxf(a0, a1), n01 = fminf(a0, a1);
        float m23 = fmaxf(a2, a3), n23 = fminf(a2, a3);
        gs[tid] = (m01 >= m23) ? (m01 + fmaxf(n01, m23)) : (m23 + fmaxf(n23, m01));
    }
    __syncthreads();

    if (tid < NGRP) {  // top-4 groups by rank (stable ties -> lower index)
        int r = 0;
        for (int j = 0; j < NGRP; ++j)
            r += (gs[j] > gs[tid]) || (gs[j] == gs[tid] && j < tid);
        gsel[tid] = (r < TOPG);
    }
    __syncthreads();

    if (tid < E_NUM) mk[tid] = gsel[tid >> 2] ? sf[tid] : -1e30f;
    __syncthreads();

    if (tid < E_NUM) {  // top-8 experts among masked
        int r = 0;
        for (int j = 0; j < E_NUM; ++j)
            r += (mk[j] > mk[tid]) || (mk[j] == mk[tid] && j < tid);
        sel[tid] = (r < TOPK) && (mk[tid] > -1e29f);
    }
    __syncthreads();

    if (tid < E_NUM) {
        float ws = 0.f;
        for (int j = 0; j < E_NUM; ++j) ws += sel[j] ? sc[j] : 0.f;
        if (sel[tid]) {
            float g = sc[tid] / ws * RSCALE;  // fold ROUTED_SCALE into gate
            int pos = atomicAdd(&cnt[tid], 1);
            perm[tid * T_TOK + pos]  = t;
            pgate[tid * T_TOK + pos] = g;
        }
    }
}

// ---------------- kernel 2: shared expert (fused up -> relu^2 -> down) ----------------
// 32-token tile per block, 8 waves. Writes the base value of `out`.

__global__ void __launch_bounds__(256)
shared_kernel(const bf16* __restrict__ xb, const float* __restrict__ Wu,
              const float* __restrict__ Wd, float* __restrict__ out) {
    extern __shared__ char smem[];
    bf16* xg  = (bf16*)smem;        // 32 x 1024 bf16 (64 KB)
    bf16* act = xg + 32 * H_DIM;    // 32 x 2048 bf16 (128 KB)

    const int tid  = threadIdx.x;
    const int wave = tid >> 5, lane = tid & 31;
    const int t0   = blockIdx.x * 32;

    {   // stage x tile into LDS (async 16B copies)
        const uint4* src = (const uint4*)(xb + (size_t)t0 * H_DIM);
        uint4* dst = (uint4*)xg;
        for (int i = tid; i < 32 * H_DIM / 8; i += 256) copy16_g2l(&dst[i], &src[i]);
        copy_fence_g2l();
    }
    __syncthreads();

    // phase 1: act[32][2048] = relu(x @ Wu^T)^2   (4 chunks of 64 cols per wave)
    for (int c = 0; c < 4; ++c) {
        const int ncol0 = wave * (SI_DIM / 8) + c * 64;
        v8f acc[2][4];
        mm_chunk4(xg, H_DIM, Wu, H_DIM, H_DIM, ncol0, lane, acc);
        const int mhi = (lane & 16) >> 1;  // +8 rows for upper half-wave
#pragma unroll
        for (int j = 0; j < 4; ++j) {
            const int n = ncol0 + 16 * j + (lane & 15);
#pragma unroll
            for (int r = 0; r < 8; ++r) {
                float v0 = fmaxf(acc[0][j][r], 0.f);
                act[(r + mhi) * SI_DIM + n] = (bf16)(v0 * v0);
                float v1 = fmaxf(acc[1][j][r], 0.f);
                act[(16 + r + mhi) * SI_DIM + n] = (bf16)(v1 * v1);
            }
        }
    }
    __syncthreads();

    // phase 2: out[32][1024] = act @ Wd^T   (2 chunks of 64 cols per wave)
    for (int c = 0; c < 2; ++c) {
        const int ncol0 = wave * (H_DIM / 8) + c * 64;
        v8f acc[2][4];
        mm_chunk4(act, SI_DIM, Wd, SI_DIM, SI_DIM, ncol0, lane, acc);
        const int mhi = (lane & 16) >> 1;
#pragma unroll
        for (int j = 0; j < 4; ++j) {
            const int n = ncol0 + 16 * j + (lane & 15);
#pragma unroll
            for (int r = 0; r < 8; ++r) {
                out[(size_t)(t0 + r + mhi) * H_DIM + n]      = acc[0][j][r];
                out[(size_t)(t0 + 16 + r + mhi) * H_DIM + n] = acc[1][j][r];
            }
        }
    }
}

// ---------------- kernel 3: routed experts (sparse, gathered by expert) ----------------
// Block = (expert e, 32-token tile of its gathered list). atomic-add into out.

__global__ void __launch_bounds__(256)
routed_kernel(const bf16* __restrict__ xb, const float* __restrict__ w_up,
              const float* __restrict__ w_down, const int* __restrict__ cnt,
              const int* __restrict__ perm, const float* __restrict__ pgate,
              float* __restrict__ out) {
    extern __shared__ char smem[];
    bf16* xg  = (bf16*)smem;       // 32 x 1024 bf16 (64 KB)
    bf16* act = xg + 32 * H_DIM;   // 32 x 512 bf16 (32 KB)
    __shared__ int   tk[32];
    __shared__ float gt[32];

    const int e    = blockIdx.x >> 7;   // /128 tiles
    const int tile = blockIdx.x & 127;
    const int base = tile * 32;
    const int ne   = cnt[e];
    if (base >= ne) return;             // uniform early-exit (before any barrier)
    int nTok = ne - base; if (nTok > 32) nTok = 32;

    const int tid  = threadIdx.x;
    const int wave = tid >> 5, lane = tid & 31;

    if (tid < 32) {
        tk[tid] = (tid < nTok) ? perm[e * T_TOK + base + tid] : 0;
        gt[tid] = (tid < nTok) ? pgate[e * T_TOK + base + tid] : 0.f;
    }
    __syncthreads();

    // gather token rows (bf16) into LDS (async 16B copies)
    for (int i = tid; i < 32 * H_DIM / 8; i += 256) {
        int m = i >> 7, col = i & 127;
        copy16_g2l(&((uint4*)xg)[i],
                   &((const uint4*)(xb + (size_t)tk[m] * H_DIM))[col]);
    }
    copy_fence_g2l();
    __syncthreads();

    // phase 1: act[32][512] = gate * relu(x @ Wu[e]^T)^2  (1 chunk of 64 cols/wave)
    const float* Wu = w_up + (size_t)e * I_DIM * H_DIM;
    {
        const int ncol0 = wave * 64;
        v8f acc[2][4];
        mm_chunk4(xg, H_DIM, Wu, H_DIM, H_DIM, ncol0, lane, acc);
        const int mhi = (lane & 16) >> 1;
#pragma unroll
        for (int j = 0; j < 4; ++j) {
            const int n = ncol0 + 16 * j + (lane & 15);
#pragma unroll
            for (int r = 0; r < 8; ++r) {
                int m0 = r + mhi;
                float v0 = fmaxf(acc[0][j][r], 0.f);
                act[m0 * I_DIM + n] = (bf16)(gt[m0] * v0 * v0);
                int m1 = 16 + r + mhi;
                float v1 = fmaxf(acc[1][j][r], 0.f);
                act[m1 * I_DIM + n] = (bf16)(gt[m1] * v1 * v1);
            }
        }
    }
    __syncthreads();

    // phase 2: out[tok] += act @ Wd[e]^T   (2 chunks of 64 cols per wave, K=512)
    const float* Wd = w_down + (size_t)e * H_DIM * I_DIM;
    for (int c = 0; c < 2; ++c) {
        const int ncol0 = wave * (H_DIM / 8) + c * 64;
        v8f acc[2][4];
        mm_chunk4(act, I_DIM, Wd, I_DIM, I_DIM, ncol0, lane, acc);
        const int mhi = (lane & 16) >> 1;
#pragma unroll
        for (int j = 0; j < 4; ++j) {
            const int n = ncol0 + 16 * j + (lane & 15);
#pragma unroll
            for (int r = 0; r < 8; ++r) {
                int m0 = r + mhi;
                if (m0 < nTok)
                    unsafeAtomicAdd(&out[(size_t)tk[m0] * H_DIM + n], acc[0][j][r]);
                int m1 = 16 + r + mhi;
                if (m1 < nTok)
                    unsafeAtomicAdd(&out[(size_t)tk[m1] * H_DIM + n], acc[1][j][r]);
            }
        }
    }
}

// ---------------- host launcher ----------------

extern "C" void kernel_launch(void* const* d_in, const int* in_sizes, int n_in,
                              void* d_out, int out_size, void* d_ws, size_t ws_size,
                              hipStream_t stream) {
    const float* x    = (const float*)d_in[0];  // [T,H]
    const float* gw   = (const float*)d_in[1];  // [E,H]
    const float* bias = (const float*)d_in[2];  // [E]
    const float* wu   = (const float*)d_in[3];  // [E,I,H]
    const float* wd   = (const float*)d_in[4];  // [E,H,I]
    const float* su   = (const float*)d_in[5];  // [SI,H]
    const float* sd   = (const float*)d_in[6];  // [H,SI]
    float* out = (float*)d_out;

    // workspace carve (~9.5 MB): xb bf16 copy of x, per-expert counts/lists
    char* p = (char*)d_ws;
    bf16* xb = (bf16*)p;           p += (size_t)T_TOK * H_DIM * sizeof(bf16);
    int* cnt = (int*)p;            p += 256;
    int* perm = (int*)p;           p += (size_t)E_NUM * T_TOK * sizeof(int);
    float* pgate = (float*)p;      p += (size_t)E_NUM * T_TOK * sizeof(float);
    (void)ws_size; (void)n_in; (void)in_sizes; (void)out_size;

    init_cnt_kernel<<<1, 64, 0, stream>>>(cnt);
    router_kernel<<<T_TOK, 256, 0, stream>>>(x, gw, bias, xb, cnt, perm, pgate);
    shared_kernel<<<T_TOK / 32, 256,
                    (32 * H_DIM + 32 * SI_DIM) * sizeof(bf16), stream>>>(xb, su, sd, out);
    routed_kernel<<<E_NUM * (T_TOK / 32), 256,
                    (32 * H_DIM + 32 * I_DIM) * sizeof(bf16), stream>>>(
        xb, wu, wd, cnt, perm, pgate, out);
}
